// GlobalBottleneck_86560770883805
// MI455X (gfx1250) — matrix operand order repaired and checked by reading, not compiled, not run on hardware
//
#include <hip/hip_runtime.h>
#include <hip/hip_bf16.h>

// GlobalBottleneck for MI455X (gfx1250): bf16 WMMA w/ fp32 accumulation.
// Pass 0: one-time weight conversion fp32 -> bf16 into workspace.
// Kernel 1: gs[BT,192](bf16) = gelu(x @ reduce_w^T + reduce_b)  (per group slice)
// Kernel 2: out = (gelu(x@expand_w^T + gs@gc_w^T + bias)) @ rf_w^T + rf_b.
//   All 160 B-fragments (160KB) staged once per block in LDS (fragment-contiguous
//   layout -> conflict-free ds_load_b128), y strip staged per-wave in LDS (64KB).
// GELU uses a branchless erf (A&S 7.1.26, |err|<=1.5e-7) built on v_rcp_f32 /
// v_exp_f32 hardware transcendentals -> no EXEC-divergent ocml expansion.

#define G_  12
#define B_  16
#define T_  2048
#define C_  64
#define R_  16
#define NG_ 192
#define H_  256
#define BT_ (B_ * T_) /* 32768 */

#define N_RW_ (G_ * R_ * C_)   /* 12288  */
#define N_EW_ (G_ * H_ * C_)   /* 196608 */
#define N_GW_ (G_ * H_ * NG_)  /* 589824 */
#define N_FW_ (G_ * C_ * H_)   /* 196608 */

// B-fragment counts in LDS: expand 16x2, gc 16x6, rf 4x8 = 160 frags of 1KB.
#define FRAG_EW_BASE_ 0
#define FRAG_GW_BASE_ 32
#define FRAG_RW_BASE_ 128
#define FRAG_TOTAL_   160

typedef __attribute__((ext_vector_type(16))) __bf16 v16bf;
typedef __attribute__((ext_vector_type(8)))  __bf16 v8bf;
typedef __attribute__((ext_vector_type(8)))  float  v8f;
typedef __attribute__((ext_vector_type(4)))  float  v4f;

// Branchless erf, Abramowitz-Stegun 7.1.26 (max abs err 1.5e-7, far below the
// bf16 matmul noise floor).  Uses hardware v_rcp_f32 / v_exp_f32; no branches,
// EXEC untouched.
__device__ __forceinline__ float erf_fast(float x) {
    const float z = __builtin_fabsf(x);
    const float t = __builtin_amdgcn_rcpf(__builtin_fmaf(0.3275911f, z, 1.0f));
    float p = __builtin_fmaf(1.061405429f, t, -1.453152027f);
    p = __builtin_fmaf(p, t, 1.421413741f);
    p = __builtin_fmaf(p, t, -0.284496736f);
    p = __builtin_fmaf(p, t, 0.254829592f);
    p *= t;
    const float e = __builtin_amdgcn_exp2f(-1.4426950408889634f * z * z);
    const float r = __builtin_fmaf(-p, e, 1.0f);
    return __builtin_copysignf(r, x);
}

__device__ __forceinline__ float gelu_exact(float v) {
    return 0.5f * v * (1.0f + erf_fast(v * 0.70710678118654752f));
}

// 16x32 bf16 A-fragment from fp32 row-major [rows x ld].
// ISA layout: lanes 0-15 -> M=lane, K = k0+{0..7, 16..23};
//             lanes 16-31 -> M=lane-16, K = k0+{8..15, 24..31}.
__device__ __forceinline__ v16bf a_frag_f32(const float* __restrict__ base, int ld,
                                            int rowbase, int k0, int lane) {
    const int m    = lane & 15;
    const int kofs = k0 + ((lane & 16) ? 8 : 0);
    const float* p = base + (size_t)(rowbase + m) * ld + kofs;
    v4f a0 = *(const v4f*)(p + 0);
    v4f a1 = *(const v4f*)(p + 4);
    v4f a2 = *(const v4f*)(p + 16);
    v4f a3 = *(const v4f*)(p + 20);
    v16bf r;
#pragma unroll
    for (int i = 0; i < 4; ++i) {
        r[i]      = (__bf16)a0[i];
        r[4 + i]  = (__bf16)a1[i];
        r[8 + i]  = (__bf16)a2[i];
        r[12 + i] = (__bf16)a3[i];
    }
    return r;
}

// Same A-fragment from a bf16 row-major source (16B-aligned chunks).
__device__ __forceinline__ v16bf a_frag_bf16(const __bf16* __restrict__ base, int ld,
                                             int rowbase, int k0, int lane) {
    const int m    = lane & 15;
    const int kofs = k0 + ((lane & 16) ? 8 : 0);
    const __bf16* p = base + (size_t)(rowbase + m) * ld + kofs;
    v8bf lo = *(const v8bf*)p;
    v8bf hi = *(const v8bf*)(p + 16);
    v16bf r;
#pragma unroll
    for (int i = 0; i < 8; ++i) { r[i] = lo[i]; r[8 + i] = hi[i]; }
    return r;
}

// 32x16 bf16 B-fragment from a pre-converted bf16 row-major weight [Ntot x Ktot]
// (weight rows are output columns): lanes 0-15 -> col n0+lane, K = k0+{0..7, 16..23};
// lanes 16-31 -> col n0+lane-16, K = k0+{8..15, 24..31}.  Two 16B loads/lane.
__device__ __forceinline__ v16bf b_frag_bf16(const __bf16* __restrict__ wbase, int ld,
                                             int n0, int k0, int lane) {
    const int n    = lane & 15;
    const int kofs = k0 + ((lane & 16) ? 8 : 0);
    const __bf16* p = wbase + (size_t)(n0 + n) * ld + kofs;
    v8bf lo = *(const v8bf*)p;
    v8bf hi = *(const v8bf*)(p + 16);
    v16bf r;
#pragma unroll
    for (int i = 0; i < 8; ++i) { r[i] = lo[i]; r[8 + i] = hi[i]; }
    return r;
}

#define WMMA_BF16(a, b, c) \
    __builtin_amdgcn_wmma_f32_16x16x32_bf16(false, (a), false, (b), (short)0, (c), false, false)

// -------- Pass 0: fp32 -> bf16 weight conversion ------------------------------
__global__ __launch_bounds__(256) void cvt_f32_to_bf16(const float* __restrict__ src,
                                                       __bf16* __restrict__ dst, int n) {
    int i = blockIdx.x * 256 + threadIdx.x;
    if (i < n) dst[i] = (__bf16)src[i];
}

// -------- Kernel 1: reduce + GELU -> gs (bf16) --------------------------------
__global__ __launch_bounds__(256) void gb_reduce_kernel(
    const float* __restrict__ x, const __bf16* __restrict__ reduce_w,
    const float* __restrict__ reduce_b, __bf16* __restrict__ gs) {
    const int lane    = threadIdx.x & 31;
    const int wave    = threadIdx.x >> 5;
    const int g       = blockIdx.y;
    const int rowbase = blockIdx.x * 128 + wave * 16;

    const float*  xg = x + (size_t)g * BT_ * C_;
    const __bf16* wg = reduce_w + (size_t)g * R_ * C_;

    v8f acc = {};
#pragma unroll
    for (int kc = 0; kc < 2; ++kc) {
        v16bf a = a_frag_f32(xg, C_, rowbase, kc * 32, lane);
        v16bf b = b_frag_bf16(wg, C_, 0, kc * 32, lane);
        acc = WMMA_BF16(a, b, acc);
    }

    // C/D layout: VGPR j holds (M = j + (lane>=16 ? 8 : 0), N = lane&15).
    const int   n    = lane & 15;
    const int   mofs = (lane & 16) ? 8 : 0;
    const float bias = reduce_b[g * R_ + n];
#pragma unroll
    for (int j = 0; j < 8; ++j) {
        float v = gelu_exact(acc[j] + bias);
        gs[(size_t)(rowbase + j + mofs) * NG_ + g * R_ + n] = (__bf16)v;
    }
}

// -------- Kernel 2: expand + global_comm + GELU + final reduce ----------------
__global__ __launch_bounds__(256) void gb_fused_kernel(
    const float* __restrict__ x, const __bf16* __restrict__ gs,
    const __bf16* __restrict__ gc_w, const float* __restrict__ gc_b,
    const __bf16* __restrict__ expand_w, const float* __restrict__ expand_b,
    const __bf16* __restrict__ rf_w, const float* __restrict__ rf_b,
    float* __restrict__ out) {
    // 160KB of staged B-fragments: blds[frag][lane] of 32B each.
    __shared__ v16bf  blds[FRAG_TOTAL_ * 32];
    // 64KB: per-wave 16x256 bf16 y strip.
    __shared__ __bf16 ylds[8 * 16 * H_];

    const int lane    = threadIdx.x & 31;
    const int wave    = threadIdx.x >> 5;
    const int g       = blockIdx.y;
    const int rowbase = blockIdx.x * 128 + wave * 16;

    const float*  xg = x + (size_t)g * BT_ * C_;
    const __bf16* ew = expand_w + (size_t)g * H_ * C_;
    const __bf16* gw = gc_w + (size_t)g * H_ * NG_;
    const __bf16* rw = rf_w + (size_t)g * C_ * H_;

    // ---- Stage all 160 B-fragments into LDS (20 per wave, uniform control) ----
#pragma unroll
    for (int i = 0; i < 20; ++i) {
        const int f = wave * 20 + i;
        v16bf frag;
        if (f < FRAG_GW_BASE_) {                       // expand_w: [ht][kc2]
            frag = b_frag_bf16(ew, C_, (f >> 1) * 16, (f & 1) * 32, lane);
        } else if (f < FRAG_RW_BASE_) {                // gc_w: [ht][kc6]
            const int q = f - FRAG_GW_BASE_;
            frag = b_frag_bf16(gw, NG_, (q / 6) * 16, (q % 6) * 32, lane);
        } else {                                       // rf_w: [nt][kc8]
            const int q = f - FRAG_RW_BASE_;
            frag = b_frag_bf16(rw, H_, (q >> 3) * 16, (q & 7) * 32, lane);
        }
        blds[f * 32 + lane] = frag;
    }

    // A-fragments held resident across all 16 h-tiles.
    v16bf xa0 = a_frag_f32(xg, C_, rowbase, 0, lane);
    v16bf xa1 = a_frag_f32(xg, C_, rowbase, 32, lane);
    v16bf ga[6];
#pragma unroll
    for (int kc = 0; kc < 6; ++kc)
        ga[kc] = a_frag_bf16(gs, NG_, rowbase, kc * 32, lane);

    __syncthreads();  // B-fragments staged

    const int n    = lane & 15;
    const int mofs = (lane & 16) ? 8 : 0;
    __bf16* yw = &ylds[wave * 16 * H_];

    // Phase 1: y = gelu(x@ew^T + gs@gw^T + bias), staged to LDS as bf16.
    for (int ht = 0; ht < 16; ++ht) {
        v8f acc = {};
        acc = WMMA_BF16(xa0, blds[(FRAG_EW_BASE_ + ht * 2 + 0) * 32 + lane], acc);
        acc = WMMA_BF16(xa1, blds[(FRAG_EW_BASE_ + ht * 2 + 1) * 32 + lane], acc);
#pragma unroll
        for (int kc = 0; kc < 6; ++kc)
            acc = WMMA_BF16(ga[kc], blds[(FRAG_GW_BASE_ + ht * 6 + kc) * 32 + lane], acc);

        const int   h    = ht * 16 + n;
        const float bias = expand_b[g * H_ + h] + gc_b[g * H_ + h];
#pragma unroll
        for (int j = 0; j < 8; ++j) {
            float v = gelu_exact(acc[j] + bias);
            yw[(j + mofs) * H_ + h] = (__bf16)v;
        }
    }

    __syncthreads();  // y strips in LDS

    // Phase 2: out[m, 0..63] = y @ rf_w^T + rf_b.  K = 256 -> 8 chunks of 32.
    v8f oacc0 = {}, oacc1 = {}, oacc2 = {}, oacc3 = {};
    for (int kc = 0; kc < 8; ++kc) {
        // y A-fragment straight from LDS (ds_load_b128 pairs).
        const int kofs = kc * 32 + ((lane & 16) ? 8 : 0);
        const __bf16* p = &yw[(lane & 15) * H_ + kofs];
        v8bf lo = *(const v8bf*)p;
        v8bf hi = *(const v8bf*)(p + 16);
        v16bf ya;
#pragma unroll
        for (int i = 0; i < 8; ++i) { ya[i] = lo[i]; ya[8 + i] = hi[i]; }

        oacc0 = WMMA_BF16(ya, blds[(FRAG_RW_BASE_ + 0 * 8 + kc) * 32 + lane], oacc0);
        oacc1 = WMMA_BF16(ya, blds[(FRAG_RW_BASE_ + 1 * 8 + kc) * 32 + lane], oacc1);
        oacc2 = WMMA_BF16(ya, blds[(FRAG_RW_BASE_ + 2 * 8 + kc) * 32 + lane], oacc2);
        oacc3 = WMMA_BF16(ya, blds[(FRAG_RW_BASE_ + 3 * 8 + kc) * 32 + lane], oacc3);
    }

    float* og = out + (size_t)g * BT_ * C_;
    v8f oaccs[4] = {oacc0, oacc1, oacc2, oacc3};
#pragma unroll
    for (int nt = 0; nt < 4; ++nt) {
        const float rb = rf_b[g * C_ + nt * 16 + n];
#pragma unroll
        for (int j = 0; j < 8; ++j)
            og[(size_t)(rowbase + j + mofs) * C_ + nt * 16 + n] = oaccs[nt][j] + rb;
    }
}

extern "C" void kernel_launch(void* const* d_in, const int* in_sizes, int n_in,
                              void* d_out, int out_size, void* d_ws, size_t ws_size,
                              hipStream_t stream) {
    const float* x        = (const float*)d_in[0];
    const float* reduce_w = (const float*)d_in[1];
    const float* reduce_b = (const float*)d_in[2];
    const float* gc_w     = (const float*)d_in[3];
    const float* gc_b     = (const float*)d_in[4];
    const float* expand_w = (const float*)d_in[5];
    const float* expand_b = (const float*)d_in[6];
    const float* rf_w     = (const float*)d_in[7];
    const float* rf_b     = (const float*)d_in[8];
    float*       out      = (float*)d_out;

    // Workspace layout (all 32B-aligned):
    //   gs  : [BT_, NG_] bf16                     (12.58 MB)
    //   wrb : reduce_w bf16  [G_,R_,C_]
    //   web : expand_w bf16  [G_,H_,C_]
    //   wgb : gc_w bf16      [G_,H_,NG_]
    //   wfb : rf_w bf16      [G_,C_,H_]
    __bf16* gs  = (__bf16*)d_ws;
    __bf16* wrb = gs + (size_t)BT_ * NG_;
    __bf16* web = wrb + N_RW_;
    __bf16* wgb = web + N_EW_;
    __bf16* wfb = wgb + N_GW_;

    cvt_f32_to_bf16<<<dim3((N_RW_ + 255) / 256), dim3(256), 0, stream>>>(reduce_w, wrb, N_RW_);
    cvt_f32_to_bf16<<<dim3((N_EW_ + 255) / 256), dim3(256), 0, stream>>>(expand_w, web, N_EW_);
    cvt_f32_to_bf16<<<dim3((N_GW_ + 255) / 256), dim3(256), 0, stream>>>(gc_w, wgb, N_GW_);
    cvt_f32_to_bf16<<<dim3((N_FW_ + 255) / 256), dim3(256), 0, stream>>>(rf_w, wfb, N_FW_);

    dim3 grid(BT_ / 128, G_);
    gb_reduce_kernel<<<grid, dim3(256), 0, stream>>>(x, wrb, reduce_b, gs);
    gb_fused_kernel<<<grid, dim3(256), 0, stream>>>(x, gs, wgb, gc_b, web,
                                                    expand_b, wfb, rf_b, out);
}
